// GraphDDPMUNet_18227841204841
// MI455X (gfx1250) — matrix operand order repaired and checked by reading, not compiled, not run on hardware
//
#include <hip/hip_runtime.h>
#include <cstdint>

// ---------------------------------------------------------------------------
// Graph U-Net (DDPM) for MI455X / gfx1250.
// Heavy GEMMs run through a bf16 WMMA kernel (v_wmma_f32_16x16x32_bf16, f32
// accumulate) with TDM (tensor_load_to_lds) staging of the A tile and
// vectorized staging of the B tile. LN/FiLM/SiLU, top-k, gather/scatter and
// tiny cond MLPs are f32.
// ---------------------------------------------------------------------------

typedef unsigned short u16;
typedef __attribute__((ext_vector_type(16))) __bf16 bf16x16;
typedef __attribute__((ext_vector_type(8)))  float  f32x8;
typedef __attribute__((ext_vector_type(4))) unsigned uint32x4;
typedef __attribute__((ext_vector_type(8))) int      int32x8;
typedef __attribute__((ext_vector_type(4))) int      int32x4;

#define NN  4096   // nodes at level 0
#define BB  4      // batch
#define HD  256    // hidden dim
#define TBK 7      // total residual blocks

__device__ __forceinline__ u16 f2bf(float f) {
  unsigned u = __float_as_uint(f);
  u = (u + 0x7FFFu + ((u >> 16) & 1u)) >> 16;   // round-to-nearest-even
  return (u16)u;
}
__device__ __forceinline__ float silu_f(float x) { return x / (1.f + __expf(-x)); }

union FragU { bf16x16 v; uint4 q[2]; };

// ---------------------------------------------------------------------------
// Aligned bf16 WMMA GEMM (M%128==0, N%64==0, K%32==0).
// Block tile 128x64x32, 256 threads = 8 waves (4x2 grid of 32x32 wave tiles,
// 4 WMMAs per wave per k-step). A tile staged by the Tensor Data Mover with
// LDS row padding baked into the descriptor; B tile staged transposed by ALU.
// ---------------------------------------------------------------------------
__global__ void k_wmma_gemm_big(const u16* __restrict__ A, long aBatch,
                                const u16* __restrict__ Bm, long bBatch,
                                float* __restrict__ C, u16* __restrict__ Cbf, long cBatch,
                                const float* __restrict__ bias,
                                int M, int N, int K, int act) {
  __shared__ __align__(16) u16 As[128][40];   // [m][k], 80B row stride (64B data + 16B pad)
  __shared__ __align__(16) u16 Bs[64][40];    // [n][k] transposed tile

  const int tid  = threadIdx.x;
  const int z    = blockIdx.z;
  const int bRow = blockIdx.y * 128;
  const int bCol = blockIdx.x * 64;

  A  += (long)z * aBatch;
  Bm += (long)z * bBatch;
  const long cOff = (long)z * cBatch;

  const int warp = tid >> 5, lane = tid & 31;
  const int wr = warp & 3, wc = warp >> 2;        // 4x2 wave grid of 32x32 tiles
  const int lrow = lane & 15, half = lane >> 4;
  const int m0 = wr * 32, n0 = wc * 32;

  f32x8 acc00, acc01, acc10, acc11;
  #pragma unroll
  for (int i = 0; i < 8; ++i) { acc00[i] = 0.f; acc01[i] = 0.f; acc10[i] = 0.f; acc11[i] = 0.f; }

  // --- TDM descriptor group 1 (constant over the k loop) ---
  // data_size=2B, pad_enable, pad after every 16 DWORDs (64B row) by 4 DWORDs (16B)
  int32x8 g1;
  g1[0] = (int)((1u << 16) | (1u << 20) | (3u << 22) | (3u << 25));
  g1[1] = (int)(((unsigned)K & 0xFFFFu) << 16);                         // tensor_dim0 lo
  g1[2] = (int)((((unsigned)K >> 16) & 0xFFFFu) |
                (((unsigned)M & 0xFFFFu) << 16));                       // tensor_dim0 hi | tensor_dim1 lo
  g1[3] = (int)((((unsigned)M >> 16) & 0xFFFFu) | (32u << 16));         // tensor_dim1 hi | tile_dim0=32
  g1[4] = (int)128u;                                                    // tile_dim1=128, tile_dim2=0
  g1[5] = (int)(unsigned)K;                                             // tensor_dim0_stride lo32
  g1[6] = 0;
  g1[7] = 0;
  int32x4 gz4; gz4[0] = 0; gz4[1] = 0; gz4[2] = 0; gz4[3] = 0;
  int32x8 gz8;
  #pragma unroll
  for (int i = 0; i < 8; ++i) gz8[i] = 0;
  const unsigned ldsA = (unsigned)(size_t)(&As[0][0]);

  for (int kt = 0; kt < K; kt += 32) {
    // ---- A tile via Tensor Data Mover (wave 0 owns the descriptor) ----
    if (warp == 0) {
      unsigned long long ga = (unsigned long long)(const char*)(A + (long)bRow * K + kt);
      uint32x4 g0;
      g0[0] = 1u;                                            // count=1 (valid user D#)
      g0[1] = ldsA;                                          // LDS byte address
      g0[2] = (unsigned)(ga & 0xFFFFFFFFu);                  // global addr lo
      g0[3] = (unsigned)((ga >> 32) & 0x01FFFFFFu) | (2u << 30);  // addr hi | type=2
      __builtin_amdgcn_tensor_load_to_lds(g0, g1, gz4, gz4, gz8, 0);
    }
    // ---- B tile (32x64) staged transposed: vectorized b128 global loads ----
    {
      const int kr = tid >> 3, c0 = (tid & 7) * 8;
      const uint4 qv = *(const uint4*)(Bm + (long)(kt + kr) * N + bCol + c0);
      const u16* pv = (const u16*)&qv;
      #pragma unroll
      for (int j = 0; j < 8; ++j) Bs[c0 + j][kr] = pv[j];
    }
    if (kt + 32 < K)
      __builtin_prefetch(Bm + (long)(kt + 32 + (tid >> 3)) * N + bCol, 0, 1);
    if (warp == 0) __builtin_amdgcn_s_wait_tensorcnt(0);
    __syncthreads();

    // ---- fragments (CDNA5 layouts, CDNA5 ISA) ----
    FragU a0, a1, b0, b1;
    a0.q[0] = *(const uint4*)&As[m0 + lrow][8 * half];
    a0.q[1] = *(const uint4*)&As[m0 + lrow][16 + 8 * half];
    a1.q[0] = *(const uint4*)&As[m0 + 16 + lrow][8 * half];
    a1.q[1] = *(const uint4*)&As[m0 + 16 + lrow][16 + 8 * half];
    b0.q[0] = *(const uint4*)&Bs[n0 + lrow][8 * half];
    b0.q[1] = *(const uint4*)&Bs[n0 + lrow][16 + 8 * half];
    b1.q[0] = *(const uint4*)&Bs[n0 + 16 + lrow][8 * half];
    b1.q[1] = *(const uint4*)&Bs[n0 + 16 + lrow][16 + 8 * half];

    acc00 = __builtin_amdgcn_wmma_f32_16x16x32_bf16(false, a0.v, false, b0.v, (short)0, acc00, false, false);
    acc01 = __builtin_amdgcn_wmma_f32_16x16x32_bf16(false, a0.v, false, b1.v, (short)0, acc01, false, false);
    acc10 = __builtin_amdgcn_wmma_f32_16x16x32_bf16(false, a1.v, false, b0.v, (short)0, acc10, false, false);
    acc11 = __builtin_amdgcn_wmma_f32_16x16x32_bf16(false, a1.v, false, b1.v, (short)0, acc11, false, false);
    __syncthreads();
  }

  // ---- epilogue (no guards: aligned dims) ----
  #pragma unroll
  for (int i = 0; i < 2; ++i) {
    const int rbase = bRow + m0 + i * 16 + 8 * half;
    #pragma unroll
    for (int j = 0; j < 2; ++j) {
      const int col = bCol + n0 + j * 16 + lrow;
      const float bv = bias ? bias[col] : 0.f;
      const f32x8 a = i ? (j ? acc11 : acc10) : (j ? acc01 : acc00);
      #pragma unroll
      for (int r = 0; r < 8; ++r) {
        float v = a[r] + bv;
        if (act) v = silu_f(v);
        const long o = cOff + (long)(rbase + r) * N + col;
        if (C)   C[o]   = v;
        if (Cbf) Cbf[o] = f2bf(v);
      }
    }
  }
}

// ---------------------------------------------------------------------------
// Generic guarded bf16 WMMA GEMM (edge shapes: K=3 pos MLP, N=32 out proj).
// Block tile 64x64x32, 8 waves, wave tile 16x32.
// ---------------------------------------------------------------------------
__global__ void k_wmma_gemm_gen(const u16* __restrict__ A, long aBatch,
                                const u16* __restrict__ Bm, long bBatch,
                                float* __restrict__ C, u16* __restrict__ Cbf, long cBatch,
                                const float* __restrict__ bias,
                                int M, int N, int K, int act) {
  __shared__ __align__(16) u16 As[64][40];
  __shared__ __align__(16) u16 Bs[64][40];

  const int tid  = threadIdx.x;
  const int z    = blockIdx.z;
  const int bRow = blockIdx.y * 64;
  const int bCol = blockIdx.x * 64;

  A  += (long)z * aBatch;
  Bm += (long)z * bBatch;
  const long cOff = (long)z * cBatch;

  const int warp = tid >> 5, lane = tid & 31;
  const int wr = warp >> 1, wc = warp & 1;
  const int lrow = lane & 15, half = lane >> 4;

  f32x8 acc0, acc1;
  #pragma unroll
  for (int i = 0; i < 8; ++i) { acc0[i] = 0.f; acc1[i] = 0.f; }

  for (int kt = 0; kt < K; kt += 32) {
    {
      const int r = tid >> 2, c0 = (tid & 3) * 8;
      const int gr = bRow + r;
      #pragma unroll
      for (int c = 0; c < 8; ++c) {
        const int gc = kt + c0 + c;
        u16 v = 0;
        if (gr < M && gc < K) v = A[(long)gr * K + gc];
        As[r][c0 + c] = v;
      }
    }
    {
      const int kr = tid >> 3, c0 = (tid & 7) * 8;
      const int gk = kt + kr;
      #pragma unroll
      for (int c = 0; c < 8; ++c) {
        const int gn = bCol + c0 + c;
        u16 v = 0;
        if (gk < K && gn < N) v = Bm[(long)gk * N + gn];
        Bs[c0 + c][kr] = v;
      }
    }
    __syncthreads();

    FragU af, b0, b1;
    af.q[0] = *(const uint4*)&As[wr * 16 + lrow][8 * half];
    af.q[1] = *(const uint4*)&As[wr * 16 + lrow][16 + 8 * half];
    b0.q[0] = *(const uint4*)&Bs[wc * 32 + lrow][8 * half];
    b0.q[1] = *(const uint4*)&Bs[wc * 32 + lrow][16 + 8 * half];
    b1.q[0] = *(const uint4*)&Bs[wc * 32 + 16 + lrow][8 * half];
    b1.q[1] = *(const uint4*)&Bs[wc * 32 + 16 + lrow][16 + 8 * half];

    acc0 = __builtin_amdgcn_wmma_f32_16x16x32_bf16(false, af.v, false, b0.v, (short)0, acc0, false, false);
    acc1 = __builtin_amdgcn_wmma_f32_16x16x32_bf16(false, af.v, false, b1.v, (short)0, acc1, false, false);
    __syncthreads();
  }

  const int rbase = bRow + wr * 16 + 8 * half;
  #pragma unroll
  for (int j = 0; j < 2; ++j) {
    const int col = bCol + wc * 32 + j * 16 + lrow;
    if (col >= N) continue;
    const float bv = bias ? bias[col] : 0.f;
    const f32x8 a = j ? acc1 : acc0;
    #pragma unroll
    for (int r = 0; r < 8; ++r) {
      const int row = rbase + r;
      if (row >= M) continue;
      float v = a[r] + bv;
      if (act) v = silu_f(v);
      const long o = cOff + (long)row * N + col;
      if (C)   C[o]   = v;
      if (Cbf) Cbf[o] = f2bf(v);
    }
  }
}

// ---------------------------------------------------------------------------
// f32 -> bf16 conversion
// ---------------------------------------------------------------------------
__global__ void k_cvt_bf16(const float* __restrict__ s, u16* __restrict__ d, long n) {
  long e = (long)blockIdx.x * blockDim.x + threadIdx.x;
  if (e < n) d[e] = f2bf(s[e]);
}

// ---------------------------------------------------------------------------
// degree + D^-1/2 for adj + I
// ---------------------------------------------------------------------------
__global__ void k_deg(const float* __restrict__ adj, float* __restrict__ dinv) {
  const int i = blockIdx.x, tid = threadIdx.x;
  __shared__ float red[256];
  float s = 0.f;
  for (int j = tid; j < NN; j += 256) s += adj[(long)i * NN + j];
  red[tid] = s; __syncthreads();
  for (int st = 128; st > 0; st >>= 1) { if (tid < st) red[tid] += red[tid + st]; __syncthreads(); }
  if (tid == 0) { float d = red[0] + 1.f; dinv[i] = (d > 0.f) ? rsqrtf(d) : 0.f; }
}

__global__ void k_build_ahat(const float* __restrict__ adj, const float* __restrict__ dinv,
                             u16* __restrict__ out) {
  long e = (long)blockIdx.x * blockDim.x + threadIdx.x;
  if (e >= (long)NN * NN) return;
  int i = (int)(e / NN), j = (int)(e % NN);
  float a = adj[e] + (i == j ? 1.f : 0.f);
  out[e] = f2bf(a * dinv[i] * dinv[j]);
}

// ---------------------------------------------------------------------------
// LayerNorm fusions (one block = one row of H=256)
// ---------------------------------------------------------------------------
__device__ __forceinline__ float blk_ln(float v, float* red, int tid, float& dcent) {
  red[tid] = v; __syncthreads();
  for (int s = 128; s > 0; s >>= 1) { if (tid < s) red[tid] += red[tid + s]; __syncthreads(); }
  const float mu = red[0] * (1.f / 256.f); __syncthreads();
  dcent = v - mu;
  red[tid] = dcent * dcent; __syncthreads();
  for (int s = 128; s > 0; s >>= 1) { if (tid < s) red[tid] += red[tid + s]; __syncthreads(); }
  const float var = red[0] * (1.f / 256.f); __syncthreads();
  return rsqrtf(var + 1e-5f);
}

__global__ void k_ln_cond_silu(const float* __restrict__ x, const float* __restrict__ g,
                               const float* __restrict__ bta, const float* __restrict__ csig,
                               u16* __restrict__ out, int n) {
  const int row = blockIdx.x, tid = threadIdx.x, b = row / n;
  __shared__ float red[256];
  float dcent;
  const float v = x[(long)row * HD + tid];
  const float rstd = blk_ln(v, red, tid, dcent);
  float y = dcent * rstd * g[tid] + bta[tid] + csig[b * HD + tid];
  out[(long)row * HD + tid] = f2bf(silu_f(y));
}

__global__ void k_ln_film_add(float* __restrict__ x, const float* __restrict__ t,
                              const float* __restrict__ g, const float* __restrict__ bta,
                              const float* __restrict__ film, int blk, int n) {
  const int row = blockIdx.x, tid = threadIdx.x, b = row / n;
  __shared__ float red[256];
  float dcent;
  const float v = t[(long)row * HD + tid];
  const float rstd = blk_ln(v, red, tid, dcent);
  const float y = dcent * rstd * g[tid] + bta[tid];
  const long fb = (((long)b * TBK + blk) * 2) * HD + tid;
  const float gamma = 1.f + film[fb];
  const float beta  = film[fb + HD];
  x[(long)row * HD + tid] += y * gamma + beta;
}

__global__ void k_ln_bf16(const float* __restrict__ x, const float* __restrict__ g,
                          const float* __restrict__ bta, u16* __restrict__ out) {
  const int row = blockIdx.x, tid = threadIdx.x;
  __shared__ float red[256];
  float dcent;
  const float v = x[(long)row * HD + tid];
  const float rstd = blk_ln(v, red, tid, dcent);
  out[(long)row * HD + tid] = f2bf(dcent * rstd * g[tid] + bta[tid]);
}

// ---------------------------------------------------------------------------
// Small f32 GEMM for tiny (M<=4) cond / FiLM chains
// ---------------------------------------------------------------------------
__global__ void k_small_gemm(const float* __restrict__ A, const float* __restrict__ W,
                             const float* __restrict__ bias, float* __restrict__ out,
                             int M, int K, int N, int act) {
  const int idx = blockIdx.x * blockDim.x + threadIdx.x;
  if (idx >= M * N) return;
  const int r = idx / N, c = idx % N;
  float s = bias ? bias[c] : 0.f;
  for (int k = 0; k < K; ++k) s += A[r * K + k] * W[(long)k * N + c];
  if (act) s = silu_f(s);
  out[idx] = s;
}

// ---------------------------------------------------------------------------
// pooling score: s[row] = dot(t128[row], w2[d]) + b2[d]
// ---------------------------------------------------------------------------
__global__ void k_score(const float* __restrict__ t, const float* __restrict__ w,
                        const float* __restrict__ b2, float* __restrict__ s, int rows) {
  const int r = blockIdx.x * blockDim.x + threadIdx.x;
  if (r >= rows) return;
  float acc = b2[0];
  for (int k = 0; k < 128; ++k) acc += t[(long)r * 128 + k] * w[k];
  s[r] = acc;
}

// ---------------------------------------------------------------------------
// per-batch bitonic top-k (n is a power of two; descending value, ascending idx)
// ---------------------------------------------------------------------------
__global__ void k_topk(const float* __restrict__ s, int n, int k, int* __restrict__ idx_out) {
  __shared__ float vals[4096];
  __shared__ int   inds[4096];
  const int b = blockIdx.x, tid = threadIdx.x;
  for (int i = tid; i < n; i += 256) { vals[i] = s[(long)b * n + i]; inds[i] = i; }
  __syncthreads();
  for (int size = 2; size <= n; size <<= 1) {
    for (int stride = size >> 1; stride > 0; stride >>= 1) {
      for (int t = tid; t < (n >> 1); t += 256) {
        const int pos = 2 * stride * (t / stride) + (t % stride);
        const int par = pos + stride;
        const bool d = ((pos & size) == 0);
        const float av = vals[pos], bv = vals[par];
        const int   ai = inds[pos], bi = inds[par];
        const bool before = (bv > av) || (bv == av && bi < ai);
        if (before == d) { vals[pos] = bv; vals[par] = av; inds[pos] = bi; inds[par] = ai; }
      }
      __syncthreads();
    }
  }
  for (int i = tid; i < k; i += 256) idx_out[(long)b * k + i] = inds[i];
}

// ---------------------------------------------------------------------------
// gather / pooled adjacency / scatter-unpool / misc elementwise
// ---------------------------------------------------------------------------
__global__ void k_gather(const float* __restrict__ h, const int* __restrict__ idx,
                         int n_old, int n_new, float* __restrict__ out) {
  long e = (long)blockIdx.x * blockDim.x + threadIdx.x;
  const long total = (long)BB * n_new * HD;
  if (e >= total) return;
  const int c = (int)(e % HD);
  const long t = e / HD;
  const int i = (int)(t % n_new), b = (int)(t / n_new);
  out[e] = h[((long)b * n_old + idx[(long)b * n_new + i]) * HD + c];
}

__global__ void k_pool_adj(const u16* __restrict__ src, long srcStride,
                           const int* __restrict__ idx, int n_old, int n_new,
                           u16* __restrict__ dst) {
  const int b = blockIdx.z;
  long e = (long)blockIdx.x * blockDim.x + threadIdx.x;
  const long tot = (long)n_new * n_new;
  if (e >= tot) return;
  const int i = (int)(e / n_new), j = (int)(e % n_new);
  const int* ix = idx + (long)b * n_new;
  dst[(long)b * tot + e] = src[(long)b * srcStride + (long)ix[i] * n_old + ix[j]];
}

__global__ void k_scatter_add(float* __restrict__ big, const float* __restrict__ hs,
                              const int* __restrict__ idx, int n_old, int n_new) {
  long e = (long)blockIdx.x * blockDim.x + threadIdx.x;
  const long total = (long)BB * n_new * HD;
  if (e >= total) return;
  const int c = (int)(e % HD);
  const long t = e / HD;
  const int i = (int)(t % n_new), b = (int)(t / n_new);
  big[((long)b * n_old + idx[(long)b * n_new + i]) * HD + c] += hs[e];
}

__global__ void k_add_pemb(float* __restrict__ h, const float* __restrict__ pemb) {
  long e = (long)blockIdx.x * blockDim.x + threadIdx.x;
  const long total = (long)BB * NN * HD;
  if (e >= total) return;
  h[e] += pemb[e % ((long)NN * HD)];
}

// ---------------------------------------------------------------------------
// host orchestration
// ---------------------------------------------------------------------------
extern "C" void kernel_launch(void* const* d_in, const int* in_sizes, int n_in,
                              void* d_out, int out_size, void* d_ws, size_t ws_size,
                              hipStream_t stream) {
  const float* x0        = (const float*)d_in[0];
  const float* adj       = (const float*)d_in[1];
  const float* cond      = (const float*)d_in[2];
  const float* pos       = (const float*)d_in[3];
  const float* in_proj_w = (const float*)d_in[5];
  const float* in_proj_b = (const float*)d_in[6];
  const float* pos_w1    = (const float*)d_in[7];
  const float* pos_b1    = (const float*)d_in[8];
  const float* pos_w2    = (const float*)d_in[9];
  const float* pos_b2    = (const float*)d_in[10];
  const float* cond_w1   = (const float*)d_in[11];
  const float* cond_b1   = (const float*)d_in[12];
  const float* cond_w2   = (const float*)d_in[13];
  const float* cond_b2   = (const float*)d_in[14];
  const float* film_w1   = (const float*)d_in[15];
  const float* film_b1   = (const float*)d_in[16];
  const float* film_w2   = (const float*)d_in[17];
  const float* film_b2   = (const float*)d_in[18];
  const float* film_w3   = (const float*)d_in[19];
  const float* film_b3   = (const float*)d_in[20];
  const float* blk_ln1_g = (const float*)d_in[21];
  const float* blk_ln1_b = (const float*)d_in[22];
  const float* blk_ln2_g = (const float*)d_in[23];
  const float* blk_ln2_b = (const float*)d_in[24];
  const float* blk_conv_w= (const float*)d_in[25];
  const float* blk_conv_b= (const float*)d_in[26];
  const float* blk_cond_w= (const float*)d_in[27];
  const float* blk_cond_b= (const float*)d_in[28];
  const float* pool_ln_g = (const float*)d_in[29];
  const float* pool_ln_b = (const float*)d_in[30];
  const float* pool_w1   = (const float*)d_in[31];
  const float* pool_b1   = (const float*)d_in[32];
  const float* pool_w2   = (const float*)d_in[33];
  const float* pool_b2   = (const float*)d_in[34];
  const float* out_ln_g  = (const float*)d_in[35];
  const float* out_ln_b  = (const float*)d_in[36];
  const float* out_w     = (const float*)d_in[37];
  const float* out_b     = (const float*)d_in[38];
  float* outp = (float*)d_out;

  char* base = (char*)d_ws;
  size_t off = 0;
  auto alloc = [&](size_t bytes) -> void* {
    off = (off + 255) & ~(size_t)255;
    void* p = base + off;
    off += bytes;
    return p;
  };

  float* dinv  = (float*)alloc((size_t)NN * 4);
  u16* Ahat0   = (u16*)alloc((size_t)NN * NN * 2);
  u16* A1      = (u16*)alloc((size_t)BB * 2048 * 2048 * 2);
  u16* A2      = (u16*)alloc((size_t)BB * 1024 * 1024 * 2);
  u16* A3      = (u16*)alloc((size_t)BB * 512 * 512 * 2);
  float* h0    = (float*)alloc((size_t)BB * 4096 * HD * 4);
  float* h1    = (float*)alloc((size_t)BB * 2048 * HD * 4);
  float* h2    = (float*)alloc((size_t)BB * 1024 * HD * 4);
  float* h3    = (float*)alloc((size_t)BB * 512 * HD * 4);
  u16* hh_bf   = (u16*)alloc((size_t)BB * 4096 * HD * 2);
  u16* sp_bf   = (u16*)alloc((size_t)BB * 4096 * HD * 2);
  u16* ln_bf   = (u16*)alloc((size_t)BB * 4096 * HD * 2);
  float* t2    = (float*)alloc((size_t)BB * 4096 * HD * 4);
  float* t128  = (float*)alloc((size_t)BB * 4096 * 128 * 4);
  float* scores= (float*)alloc((size_t)BB * 4096 * 4);
  int* idx0    = (int*)alloc((size_t)BB * 2048 * 4);
  int* idx1    = (int*)alloc((size_t)BB * 1024 * 4);
  int* idx2    = (int*)alloc((size_t)BB * 512 * 4);
  u16* x0_bf   = (u16*)alloc((size_t)BB * NN * 32 * 2);
  u16* pos_bf  = (u16*)alloc((size_t)NN * 3 * 2);
  u16* tpos_bf = (u16*)alloc((size_t)NN * HD * 2);
  float* pemb  = (float*)alloc((size_t)NN * HD * 4);
  u16* ipw_bf  = (u16*)alloc((size_t)32 * HD * 2);
  u16* pw1_bf  = (u16*)alloc((size_t)3 * HD * 2);
  u16* pw2_bf  = (u16*)alloc((size_t)HD * HD * 2);
  u16* cw_bf   = (u16*)alloc((size_t)TBK * HD * HD * 2);
  u16* plw1_bf = (u16*)alloc((size_t)3 * HD * 128 * 2);
  u16* ow_bf   = (u16*)alloc((size_t)HD * 32 * 2);
  float* c1    = (float*)alloc((size_t)BB * HD * 4);
  float* cemb  = (float*)alloc((size_t)BB * HD * 4);
  float* f1    = (float*)alloc((size_t)BB * 512 * 4);
  float* f2    = (float*)alloc((size_t)BB * 512 * 4);
  float* film  = (float*)alloc((size_t)BB * TBK * 2 * HD * 4);
  float* csig  = (float*)alloc((size_t)TBK * BB * HD * 4);

  auto cvt = [&](const float* s, u16* d, long n) {
    k_cvt_bf16<<<dim3((unsigned)((n + 255) / 256)), 256, 0, stream>>>(s, d, n);
  };
  // routes to the TDM/WMMA fast kernel when the shape is tile-aligned
  auto gemm = [&](const u16* A, long aB, const u16* Bm, long bB,
                  float* C, u16* Cbf, long cB, const float* bias,
                  int M, int N_, int K, int act, int Z) {
    if ((M % 128 == 0) && (N_ % 64 == 0) && (K % 32 == 0)) {
      dim3 g(N_ / 64, M / 128, Z);
      k_wmma_gemm_big<<<g, 256, 0, stream>>>(A, aB, Bm, bB, C, Cbf, cB, bias, M, N_, K, act);
    } else {
      dim3 g((N_ + 63) / 64, (M + 63) / 64, Z);
      k_wmma_gemm_gen<<<g, 256, 0, stream>>>(A, aB, Bm, bB, C, Cbf, cB, bias, M, N_, K, act);
    }
  };

  // ---- weight/input conversions to bf16 ----
  cvt(x0, x0_bf, (long)BB * NN * 32);
  cvt(pos, pos_bf, (long)NN * 3);
  cvt(in_proj_w, ipw_bf, 32L * HD);
  cvt(pos_w1, pw1_bf, 3L * HD);
  cvt(pos_w2, pw2_bf, (long)HD * HD);
  cvt(blk_conv_w, cw_bf, (long)TBK * HD * HD);
  cvt(pool_w1, plw1_bf, 3L * HD * 128);
  cvt(out_w, ow_bf, (long)HD * 32);

  // ---- normalized adjacency ----
  k_deg<<<NN, 256, 0, stream>>>(adj, dinv);
  k_build_ahat<<<(unsigned)(((long)NN * NN + 255) / 256), 256, 0, stream>>>(adj, dinv, Ahat0);

  // ---- position MLP: pemb = silu(pos@w1+b1)@w2 + b2 ----
  gemm(pos_bf, 0, pw1_bf, 0, nullptr, tpos_bf, 0, pos_b1, NN, HD, 3, 1, 1);
  gemm(tpos_bf, 0, pw2_bf, 0, pemb, nullptr, 0, pos_b2, NN, HD, HD, 0, 1);

  // ---- input projection + pemb ----
  gemm(x0_bf, 0, ipw_bf, 0, h0, nullptr, 0, in_proj_b, BB * NN, HD, 32, 0, 1);
  k_add_pemb<<<(unsigned)(((long)BB * NN * HD + 255) / 256), 256, 0, stream>>>(h0, pemb);

  // ---- cond / FiLM chains (tiny, f32) ----
  auto smallg = [&](const float* A, const float* W, const float* bias, float* out,
                    int M, int K, int N_, int act) {
    k_small_gemm<<<(M * N_ + 255) / 256, 256, 0, stream>>>(A, W, bias, out, M, K, N_, act);
  };
  smallg(cond, cond_w1, cond_b1, c1, BB, HD, HD, 1);
  smallg(c1, cond_w2, cond_b2, cemb, BB, HD, HD, 0);
  smallg(cond, film_w1, film_b1, f1, BB, HD, 512, 1);
  smallg(f1, film_w2, film_b2, f2, BB, 512, 512, 1);
  smallg(f2, film_w3, film_b3, film, BB, 512, TBK * 2 * HD, 0);
  for (int i = 0; i < TBK; ++i)
    smallg(cemb, blk_cond_w + (long)i * HD * HD, blk_cond_b + (long)i * HD,
           csig + (long)i * BB * HD, BB, HD, HD, 0);

  // ---- residual block ----
  auto res_block = [&](float* h, const u16* Ab, long aStride, int n, int i) {
    k_ln_cond_silu<<<BB * n, 256, 0, stream>>>(h, blk_ln1_g + (long)i * HD,
                                               blk_ln1_b + (long)i * HD,
                                               csig + (long)i * BB * HD, hh_bf, n);
    gemm(Ab, aStride, hh_bf, (long)n * HD, nullptr, sp_bf, (long)n * HD,
         nullptr, n, HD, n, 0, BB);
    gemm(sp_bf, 0, cw_bf + (long)i * HD * HD, 0, t2, nullptr, 0,
         blk_conv_b + (long)i * HD, BB * n, HD, HD, 0, 1);
    k_ln_film_add<<<BB * n, 256, 0, stream>>>(h, t2, blk_ln2_g + (long)i * HD,
                                              blk_ln2_b + (long)i * HD, film, i, n);
  };

  float* hbuf[4] = { h0, h1, h2, h3 };
  u16*   Abuf[4] = { Ahat0, A1, A2, A3 };
  long   aStr[4] = { 0, 2048L * 2048, 1024L * 1024, 512L * 512 };
  int    ns[4]   = { 4096, 2048, 1024, 512 };
  int*   idxs[3] = { idx0, idx1, idx2 };

  int blk = 0;
  for (int d = 0; d < 3; ++d) {
    const int n = ns[d], k = ns[d + 1];
    res_block(hbuf[d], Abuf[d], aStr[d], n, blk); blk++;
    k_ln_bf16<<<BB * n, 256, 0, stream>>>(hbuf[d], pool_ln_g + (long)d * HD,
                                          pool_ln_b + (long)d * HD, ln_bf);
    gemm(ln_bf, 0, plw1_bf + (long)d * HD * 128, 0, t128, nullptr, 0,
         pool_b1 + (long)d * 128, BB * n, 128, HD, 1, 1);
    k_score<<<(BB * n + 255) / 256, 256, 0, stream>>>(t128, pool_w2 + (long)d * 128,
                                                      pool_b2 + d, scores, BB * n);
    k_topk<<<BB, 256, 0, stream>>>(scores, n, k, idxs[d]);
    k_gather<<<(unsigned)(((long)BB * k * HD + 255) / 256), 256, 0, stream>>>(
        hbuf[d], idxs[d], n, k, hbuf[d + 1]);
    {
      dim3 g((unsigned)(((long)k * k + 255) / 256), 1, BB);
      k_pool_adj<<<g, 256, 0, stream>>>(Abuf[d], aStr[d], idxs[d], n, k, Abuf[d + 1]);
    }
  }
  res_block(hbuf[3], Abuf[3], aStr[3], ns[3], blk); blk++;
  for (int d = 2; d >= 0; --d) {
    const int n = ns[d], k = ns[d + 1];
    k_scatter_add<<<(unsigned)(((long)BB * k * HD + 255) / 256), 256, 0, stream>>>(
        hbuf[d], hbuf[d + 1], idxs[d], n, k);
    res_block(hbuf[d], Abuf[d], aStr[d], n, blk); blk++;
  }

  k_ln_bf16<<<BB * NN, 256, 0, stream>>>(h0, out_ln_g, out_ln_b, ln_bf);
  gemm(ln_bf, 0, ow_bf, 0, outp, nullptr, 0, out_b, BB * NN, 32, HD, 0, 1);
}
